// Item_GraphConvolution_mid_attention_16140487098644
// MI455X (gfx1250) — matrix operand order repaired and checked by reading, not compiled, not run on hardware
//
#include <hip/hip_runtime.h>
#include <hip/hip_bf16.h>

typedef __attribute__((ext_vector_type(16))) _Float16 v16h;
typedef __attribute__((ext_vector_type(8)))  _Float16 v8h;
typedef __attribute__((ext_vector_type(8)))  float    v8f;

#define ALPHA 0.2f
#define WT_STRIDE 264   // padded K-stride (halfs) for transposed W in LDS

// ---------------------------------------------------------------------------
// GEMM1: support = relu(feature @ W)
// feature [M,256] row-major, W [256,128] row-major, support [M,128] row-major.
// W is staged once per block into LDS, transposed + converted to f16 in the
// WMMA B-fragment layout: Wt[n][k] (row stride WT_STRIDE halfs).
// One wave computes a 16(M) x 128(N) strip; B frags are 2x ds_load_b128.
// ---------------------------------------------------------------------------
__global__ __launch_bounds__(256) void gemm1_relu_kernel(
    const float* __restrict__ feat, const float* __restrict__ W,
    float* __restrict__ support, int M) {
  __shared__ _Float16 Wt[128 * WT_STRIDE];   // ~67.6 KB

  // Cooperative stage: Wt[n][k] = (f16) W[k][n]   (coalesced global reads)
  for (int i = 0; i < 128; ++i) {
    const int idx = i * 256 + threadIdx.x;   // idx over W: k = idx>>7, n = idx&127
    const int k = idx >> 7;
    const int n = idx & 127;
    Wt[n * WT_STRIDE + k] = (_Float16)W[idx];
  }
  __syncthreads();

  const int lane = threadIdx.x & 31;
  const int wave = threadIdx.x >> 5;
  const int mt   = blockIdx.x * 8 + wave;
  if (mt * 16 < M) {                         // wave-uniform guard: EXEC full
    const int m0  = mt * 16;
    const int nl  = lane & 15;               // N index / A-row index
    const int sub = lane >> 4;               // lane half

    v8f acc[8];
    const v8f vz = {0.f, 0.f, 0.f, 0.f, 0.f, 0.f, 0.f, 0.f};
#pragma unroll
    for (int i = 0; i < 8; ++i) acc[i] = vz;

    const float* arow = feat + (size_t)(m0 + nl) * 256;

    for (int k0 = 0; k0 < 256; k0 += 32) {
      // A fragment (16x32 f16): two contiguous 8-float chunks per lane
      v16h a;
      const float4* p1 = (const float4*)(arow + k0 + 16 * sub);
      const float4* p2 = (const float4*)(arow + k0 + 16 * sub + 8);
      const float4 a0 = p1[0], a1 = p1[1], a2 = p2[0], a3 = p2[1];
      a[0]  = (_Float16)a0.x; a[1]  = (_Float16)a0.y;
      a[2]  = (_Float16)a0.z; a[3]  = (_Float16)a0.w;
      a[4]  = (_Float16)a1.x; a[5]  = (_Float16)a1.y;
      a[6]  = (_Float16)a1.z; a[7]  = (_Float16)a1.w;
      a[8]  = (_Float16)a2.x; a[9]  = (_Float16)a2.y;
      a[10] = (_Float16)a2.z; a[11] = (_Float16)a2.w;
      a[12] = (_Float16)a3.x; a[13] = (_Float16)a3.y;
      a[14] = (_Float16)a3.z; a[15] = (_Float16)a3.w;

#pragma unroll
      for (int nt = 0; nt < 8; ++nt) {
        const int n0 = nt * 16;
        // B fragment: lane = column (n0+nl), 16 contiguous K halfs from LDS
        const _Float16* wp = &Wt[(n0 + nl) * WT_STRIDE + k0 + 16 * sub];
        const v8h blo = *(const v8h*)(wp);       // ds_load_b128
        const v8h bhi = *(const v8h*)(wp + 8);   // ds_load_b128
        v16h bf;
#pragma unroll
        for (int h = 0; h < 8; ++h) { bf[h] = blo[h]; bf[8 + h] = bhi[h]; }
        acc[nt] = __builtin_amdgcn_wmma_f32_16x16x32_f16(
            false, a, false, bf, (short)0, acc[nt], false, false);
      }
    }

    // Store with ReLU. D layout: VGPR r -> row m0 + r + 8*sub, col n0 + nl
#pragma unroll
    for (int nt = 0; nt < 8; ++nt) {
#pragma unroll
      for (int r = 0; r < 8; ++r) {
        float v = acc[nt][r];
        v = v > 0.f ? v : 0.f;
        support[(size_t)(m0 + r + 8 * sub) * 128 + nt * 16 + nl] = v;
      }
    }
  }
}

// ---------------------------------------------------------------------------
// SPMM: y[row] += val * x[col] over COO edges. One wave per edge, one float4
// per lane (512B coalesced gather), 4 f32 atomics per lane (L2 RMW).
// ---------------------------------------------------------------------------
__global__ __launch_bounds__(256) void spmm_atomic_kernel(
    const int* __restrict__ rows, const int* __restrict__ cols,
    const float* __restrict__ vals, const float* __restrict__ x,
    float* __restrict__ y, int nE) {
  const long long t = (long long)blockIdx.x * 256 + threadIdx.x;
  const long long e = t >> 5;
  if (e >= nE) return;
  const int lane = (int)(t & 31);
  const int r = rows[e];
  const int c = cols[e];
  const float v = vals[e];
  const float4 s = *(const float4*)(x + (size_t)c * 128 + lane * 4);
  float* dst = y + (size_t)r * 128 + lane * 4;
  atomicAdd(dst + 0, v * s.x);
  atomicAdd(dst + 1, v * s.y);
  atomicAdd(dst + 2, v * s.z);
  atomicAdd(dst + 3, v * s.w);
}

// ---------------------------------------------------------------------------
// GEMM2: out = leaky_relu((s1+sup) @ cat_W[:, :128]^T
//                       + (s2-sup) @ cat_W[:, 128:]^T + cat_b) + b
// cat_W [128,256] row-major is already B-major (B[k][j] = cat_W[j][k]);
// staged to LDS as f16 with a fully coalesced copy.
// ---------------------------------------------------------------------------
__global__ __launch_bounds__(256) void gemm2_kernel(
    const float* __restrict__ s1, const float* __restrict__ s2,
    const float* __restrict__ sup, const float* __restrict__ catW,
    const float* __restrict__ catb, const float* __restrict__ bvec,
    float* __restrict__ out, int M) {
  __shared__ _Float16 Cw[128 * 256];   // 64 KB, Cw[j][k]

  for (int i = 0; i < 128; ++i) {
    const int idx = i * 256 + threadIdx.x;
    Cw[idx] = (_Float16)catW[idx];     // coalesced global + LDS
  }
  __syncthreads();

  const int lane = threadIdx.x & 31;
  const int wave = threadIdx.x >> 5;
  const int mt   = blockIdx.x * 8 + wave;
  if (mt * 16 < M) {
    const int m0  = mt * 16;
    const int nl  = lane & 15;
    const int sub = lane >> 4;

    v8f acc[8];
    const v8f vz = {0.f, 0.f, 0.f, 0.f, 0.f, 0.f, 0.f, 0.f};
#pragma unroll
    for (int i = 0; i < 8; ++i) acc[i] = vz;

    const size_t arow = (size_t)(m0 + nl) * 128;
    const float* supr = sup + arow;

#pragma unroll
    for (int phase = 0; phase < 2; ++phase) {
      const float* srow = (phase == 0 ? s1 : s2) + arow;
      const float  sgn  = (phase == 0) ? 1.f : -1.f;
      const int    kb   = phase * 128;
      for (int k0 = 0; k0 < 128; k0 += 32) {
        v16h a;
        const float4* q1 = (const float4*)(srow + k0 + 16 * sub);
        const float4* q2 = (const float4*)(srow + k0 + 16 * sub + 8);
        const float4* u1 = (const float4*)(supr + k0 + 16 * sub);
        const float4* u2 = (const float4*)(supr + k0 + 16 * sub + 8);
        const float4 s0 = q1[0], s1v = q1[1], s2v = q2[0], s3 = q2[1];
        const float4 t0 = u1[0], t1 = u1[1], t2 = u2[0], t3 = u2[1];
        a[0]  = (_Float16)(s0.x + sgn * t0.x);
        a[1]  = (_Float16)(s0.y + sgn * t0.y);
        a[2]  = (_Float16)(s0.z + sgn * t0.z);
        a[3]  = (_Float16)(s0.w + sgn * t0.w);
        a[4]  = (_Float16)(s1v.x + sgn * t1.x);
        a[5]  = (_Float16)(s1v.y + sgn * t1.y);
        a[6]  = (_Float16)(s1v.z + sgn * t1.z);
        a[7]  = (_Float16)(s1v.w + sgn * t1.w);
        a[8]  = (_Float16)(s2v.x + sgn * t2.x);
        a[9]  = (_Float16)(s2v.y + sgn * t2.y);
        a[10] = (_Float16)(s2v.z + sgn * t2.z);
        a[11] = (_Float16)(s2v.w + sgn * t2.w);
        a[12] = (_Float16)(s3.x + sgn * t3.x);
        a[13] = (_Float16)(s3.y + sgn * t3.y);
        a[14] = (_Float16)(s3.z + sgn * t3.z);
        a[15] = (_Float16)(s3.w + sgn * t3.w);

#pragma unroll
        for (int nt = 0; nt < 8; ++nt) {
          const int n0 = nt * 16;
          const _Float16* wp = &Cw[(n0 + nl) * 256 + kb + k0 + 16 * sub];
          const v8h blo = *(const v8h*)(wp);
          const v8h bhi = *(const v8h*)(wp + 8);
          v16h bf;
#pragma unroll
          for (int h = 0; h < 8; ++h) { bf[h] = blo[h]; bf[8 + h] = bhi[h]; }
          acc[nt] = __builtin_amdgcn_wmma_f32_16x16x32_f16(
              false, a, false, bf, (short)0, acc[nt], false, false);
        }
      }
    }

    // Epilogue: + cat_b, leaky relu, + b
#pragma unroll
    for (int nt = 0; nt < 8; ++nt) {
      const int j = nt * 16 + nl;
      const float cb = catb[j];
      const float bb = bvec[j];
#pragma unroll
      for (int r = 0; r < 8; ++r) {
        float y = acc[nt][r] + cb;
        y = y > 0.f ? y : ALPHA * y;
        out[(size_t)(m0 + r + 8 * sub) * 128 + j] = y + bb;
      }
    }
  }
}

extern "C" void kernel_launch(void* const* d_in, const int* in_sizes, int n_in,
                              void* d_out, int out_size, void* d_ws, size_t ws_size,
                              hipStream_t stream) {
  const float* feature = (const float*)d_in[0];
  const int*   erow    = (const int*)d_in[1];
  const int*   ecol    = (const int*)d_in[2];
  const float* eval    = (const float*)d_in[3];
  const float* W       = (const float*)d_in[4];
  const float* b       = (const float*)d_in[5];
  const float* catW    = (const float*)d_in[6];
  const float* catb    = (const float*)d_in[7];
  float*       out     = (float*)d_out;

  const int N  = in_sizes[0] / 256;   // nodes
  const int nE = in_sizes[1];         // edges

  const size_t mat = (size_t)N * 128; // elements per node matrix
  float* support = (float*)d_ws;
  float* s1      = support + mat;
  float* s2      = s1 + mat;

  // zero the two scatter targets every call (deterministic)
  hipMemsetAsync(s1, 0, 2 * mat * sizeof(float), stream);

  const int mtiles = (N + 15) / 16;
  const int gblk   = (mtiles + 7) / 8;
  dim3 blk(256);

  gemm1_relu_kernel<<<gblk, blk, 0, stream>>>(feature, W, support, N);

  const long long threads = (long long)nE * 32;
  const int spmmBlocks = (int)((threads + 255) / 256);
  spmm_atomic_kernel<<<spmmBlocks, blk, 0, stream>>>(erow, ecol, eval, support, s1, nE);
  spmm_atomic_kernel<<<spmmBlocks, blk, 0, stream>>>(erow, ecol, eval, s1, s2, nE);

  gemm2_kernel<<<gblk, blk, 0, stream>>>(s1, s2, support, catW, catb, b, out, N);
}
